// VQImageAutoregressiveAutoencoder_34789235098370
// MI455X (gfx1250) — matrix operand order repaired and checked by reading, not compiled, not run on hardware
//
#include <hip/hip_runtime.h>
#include <math.h>

// ---------------------------------------------------------------------------
// VQ image autoregressive autoencoder forward pass for MI455X (gfx1250).
// All large GEMMs run through v_wmma_f32_16x16x32_bf16 (wave32, f32 accum).
// B operands are always stored [N][K] (K-contiguous) so staging is a single
// 16B load + ds_store_b128 and fragments load as 2x ds_load_b128.
// Double-buffered LDS: one barrier per K-step, staging overlaps WMMA.
// ---------------------------------------------------------------------------

typedef __attribute__((ext_vector_type(16))) __bf16 bf16x16;
typedef __attribute__((ext_vector_type(8)))  float  fx8;

#define BB      32
#define NTOK    256
#define MTOT    8192            // B * N tokens
#define DMODEL  512
#define LCB     8192
#define FFD     2048
#define NHEAD   8
#define DHEAD   64
#define NLAYER  6
#define PIX     2097152         // 32*256*256

__device__ __forceinline__ __bf16 f2bf(float f) {
  unsigned u = __float_as_uint(f);
  unsigned r = (u + 0x7fffu + ((u >> 16) & 1u)) >> 16;
  unsigned short s = (unsigned short)r;
  __bf16 b; __builtin_memcpy(&b, &s, 2);
  return b;
}

__device__ __forceinline__ unsigned long long shfl_xor_u64(unsigned long long v, int m) {
  unsigned lo = (unsigned)v, hi = (unsigned)(v >> 32);
  lo = __shfl_xor(lo, m, 32);
  hi = __shfl_xor(hi, m, 32);
  return (((unsigned long long)hi) << 32) | lo;
}

__device__ __forceinline__ float gelu_tanh(float x) {
  return 0.5f * x * (1.0f + tanhf(0.7978845608028654f * (x + 0.044715f * x * x * x)));
}

// Load a 16-element bf16 fragment as two contiguous 16B chunks (ds_load_b128).
__device__ __forceinline__ bf16x16 load_frag(const __bf16* base) {
  uint4 q0 = *(const uint4*)(base);
  uint4 q1 = *(const uint4*)(base + 16);
  bf16x16 r;
  __builtin_memcpy(&r, &q0, 16);
  __builtin_memcpy((char*)&r + 16, &q1, 16);
  return r;
}

// ---------------------------------------------------------------------------
// Generic bf16 WMMA GEMM: block tile 128x64, K step 32, 8 waves (wave32),
// each wave owns a 32x32 quadrant = 2x2 tiles of 16x16.
// A is [M][K] row-major; B is [N][K] row-major (i.e. original B transposed).
// Batched via blockIdx.z with two-level (outer/inner) strides.
// ---------------------------------------------------------------------------
struct GemmParams {
  const __bf16* A; long lda, ao, ai;
  const __bf16* B; long ldb, bo, bi;
  int K;
  int ic;   // inner batch count for z = zo*ic + zi
};

struct StoreEpi {
  float*  outF;
  __bf16* outB;
  const float* bias;      // per output column, or null
  const float* residual;  // same layout as out, or null
  long ldc, co, ci;
  float alpha;
  int causal, gelu;
  __device__ void operator()(int m, int n, float v, long zo, long zi, int lane) const {
    (void)lane;
    v *= alpha;
    if (causal && n > m) v = -3.4028235e38f;
    if (bias) v += bias[n];
    if (gelu) v = gelu_tanh(v);
    long off = zo * co + zi * ci + (long)m * ldc + n;
    if (residual) v += residual[off];
    if (outF) outF[off] = v;
    if (outB) outB[off] = f2bf(v);
  }
};

struct ArgminEpi {
  const float* rn;          // row norms^2
  const float* cn;          // col norms^2
  unsigned long long* keys; // per-row packed (orderable(d2) << 32 | col)
  __device__ void operator()(int m, int n, float v, long, long, int lane) const {
    float d2 = rn[m] + cn[n] - 2.0f * v;
    unsigned u = __float_as_uint(d2);
    u = (u & 0x80000000u) ? ~u : (u | 0x80000000u);
    unsigned long long key = (((unsigned long long)u) << 32) | (unsigned)n;
#pragma unroll
    for (int s = 1; s < 16; s <<= 1) {
      unsigned long long o = shfl_xor_u64(key, s);
      if (o < key) key = o;
    }
    if ((lane & 15) == 0) atomicMin(keys + m, key);
  }
};

struct CeEpi {
  const float* rn;
  const float* cn;
  const unsigned long long* pk; // per-row argmin key (gives max logit)
  float* sumexp;
  __device__ void operator()(int m, int n, float v, long, long, int lane) const {
    float d2 = rn[m] + cn[n] - 2.0f * v;
    float logit = -sqrtf(fmaxf(d2, 1e-12f));
    unsigned u = (unsigned)(pk[m] >> 32);
    u = (u & 0x80000000u) ? (u & 0x7fffffffu) : ~u;
    float ml = -sqrtf(fmaxf(__uint_as_float(u), 1e-12f));
    float e = __expf(logit - ml);
#pragma unroll
    for (int s = 1; s < 16; s <<= 1) e += __shfl_xor(e, s, 32);
    if ((lane & 15) == 0) atomicAdd(sumexp + m, e);
  }
};

__device__ __forceinline__ void stage_tiles(__bf16 (*As)[40], __bf16 (*Bs)[40],
                                            const __bf16* A, const __bf16* B,
                                            long lda, long ldb,
                                            int row0, int col0, int k0, int t) {
#pragma unroll
  for (int i = 0; i < 2; ++i) {
    int idx = t * 2 + i;
    int r   = idx >> 2;
    int c4  = idx & 3;
    uint4 v = *(const uint4*)(A + (long)(row0 + r) * lda + k0 + c4 * 8);
    *(uint4*)(&As[r][c4 * 8]) = v;
  }
  {
    int n  = t & 63;
    int ks = (t >> 6) * 8;
    uint4 v = *(const uint4*)(B + (long)(col0 + n) * ldb + k0 + ks);
    *(uint4*)(&Bs[n][ks]) = v;
  }
}

template <typename Epi>
__global__ __launch_bounds__(256)
void gemm_bf16_kernel(GemmParams p, Epi epi) {
  // Double-buffered tiles, K-contiguous rows => fragments are 2x ds_load_b128.
  __shared__ __bf16 As[2][128][40];   // [m][k]  row stride 80B (16B aligned)
  __shared__ __bf16 Bs[2][64][40];    // [n][k]  row stride 80B (16B aligned)

  const int t    = threadIdx.x;
  const int lane = t & 31;
  const int wid  = t >> 5;
  const int wm   = wid >> 1;       // 0..3  (rows)
  const int wn   = wid & 1;        // 0..1  (cols)
  const int row0 = blockIdx.y * 128;
  const int col0 = blockIdx.x * 64;
  const long zo  = blockIdx.z / p.ic;
  const long zi  = blockIdx.z % p.ic;

  const __bf16* A = p.A + zo * p.ao + zi * p.ai;
  const __bf16* B = p.B + zo * p.bo + zi * p.bi;

  fx8 acc[2][2] = {};

  stage_tiles(As[0], Bs[0], A, B, p.lda, p.ldb, row0, col0, 0, t);
  __syncthreads();

  int buf = 0;
  for (int k0 = 0; k0 < p.K; k0 += 32) {
    // 1) fragment loads from current buffer (ds_load_b128 x2 each)
    const int ml = lane & 15;
    const int kb = (lane >> 4) << 3;
    bf16x16 afr[2], bfr[2];
#pragma unroll
    for (int tm = 0; tm < 2; ++tm)
      afr[tm] = load_frag(&As[buf][wm * 32 + tm * 16 + ml][kb]);
#pragma unroll
    for (int tn = 0; tn < 2; ++tn)
      bfr[tn] = load_frag(&Bs[buf][wn * 32 + tn * 16 + ml][kb]);

    // 2) stage next K tile into the other buffer; prefetch one tile further
    if (k0 + 32 < p.K) {
      stage_tiles(As[buf ^ 1], Bs[buf ^ 1], A, B, p.lda, p.ldb, row0, col0, k0 + 32, t);
      if (k0 + 64 < p.K) {
        int r = (t * 2) >> 2, c4 = (t * 2) & 3;
        __builtin_prefetch(A + (long)(row0 + r) * p.lda + (k0 + 64) + c4 * 8, 0, 1);
        __builtin_prefetch(B + (long)(col0 + (t & 63)) * p.ldb + (k0 + 64) + (t >> 6) * 8, 0, 1);
      }
    }

    // 3) WMMA
#pragma unroll
    for (int tm = 0; tm < 2; ++tm)
#pragma unroll
      for (int tn = 0; tn < 2; ++tn)
        acc[tm][tn] = __builtin_amdgcn_wmma_f32_16x16x32_bf16(
            false, afr[tm], false, bfr[tn], (short)0, acc[tm][tn], false, false);

    __syncthreads();
    buf ^= 1;
  }

  // --- epilogue: C/D layout = 8 VGPRs, lanes 0-15: M=r, lanes 16-31: M=r+8 ---
#pragma unroll
  for (int tm = 0; tm < 2; ++tm)
#pragma unroll
    for (int tn = 0; tn < 2; ++tn)
#pragma unroll
      for (int r = 0; r < 8; ++r) {
        int m = row0 + wm * 32 + tm * 16 + r + 8 * (lane >> 4);
        int n = col0 + wn * 32 + tn * 16 + (lane & 15);
        epi(m, n, acc[tm][tn][r], zo, zi, lane);
      }
}

// ---------------------------------------------------------------------------
// Elementwise / row-reduction kernels
// ---------------------------------------------------------------------------
__global__ __launch_bounds__(256) void f32_to_bf16_kernel(const float* s, __bf16* d, long n) {
  long i = (long)blockIdx.x * 256 + threadIdx.x;
  if (i < n) d[i] = f2bf(s[i]);
}

// Transposed convert: src is L x [K][N] f32, dst is L x [N][K] bf16.
__global__ __launch_bounds__(256) void f32_to_bf16_t_kernel(const float* src, __bf16* dst,
                                                            int K, int N, long total) {
  long i = (long)blockIdx.x * 256 + threadIdx.x;
  if (i >= total) return;
  long kn  = (long)K * N;
  long mat = i / kn;
  long r   = i % kn;
  int  n   = (int)(r / K);
  int  k   = (int)(r % K);
  dst[mat * kn + (long)n * K + k] = f2bf(src[mat * kn + (long)k * N + n]);
}

__global__ __launch_bounds__(256) void zero_f32_kernel(float* p, long n) {
  long i = (long)blockIdx.x * 256 + threadIdx.x;
  if (i < n) p[i] = 0.0f;
}

__global__ __launch_bounds__(256) void init_u64_kernel(unsigned long long* p, long n) {
  long i = (long)blockIdx.x * 256 + threadIdx.x;
  if (i < n) p[i] = 0xFFFFFFFFFFFFFFFFull;
}

__global__ __launch_bounds__(256) void patchify_kernel(const float* img, __bf16* patches) {
  long i = (long)blockIdx.x * 256 + threadIdx.x;
  if (i >= (long)MTOT * 256) return;
  int d  = (int)(i & 255);
  long m = i >> 8;
  int n  = (int)(m & 255);
  int b  = (int)(m >> 8);
  int py = d >> 4, px = d & 15;
  int hy = n >> 4, hx = n & 15;
  float v = img[(long)b * 65536 + (long)(hy * 16 + py) * 256 + hx * 16 + px];
  patches[i] = f2bf(v * 2.0f - 1.0f);
}

__global__ __launch_bounds__(256) void rownorm_kernel(const float* x, float* norms, int D) {
  __shared__ float red[256];
  long row = blockIdx.x;
  float s = 0.0f;
  for (int d = threadIdx.x; d < D; d += 256) { float v = x[row * D + d]; s += v * v; }
  red[threadIdx.x] = s; __syncthreads();
  for (int st = 128; st > 0; st >>= 1) {
    if (threadIdx.x < st) red[threadIdx.x] += red[threadIdx.x + st];
    __syncthreads();
  }
  if (threadIdx.x == 0) norms[row] = red[0];
}

__global__ __launch_bounds__(256) void gather_cbc_kernel(const unsigned long long* keys,
                                                         const float* cb, float* cbc) {
  long i = (long)blockIdx.x * 256 + threadIdx.x;
  if (i >= (long)MTOT * DMODEL) return;
  long m = i >> 9;               // /512
  int  d = (int)(i & 511);
  unsigned code = (unsigned)(keys[m] & 0xFFFFFFFFu);
  cbc[i] = cb[(long)code * DMODEL + d];
}

// rotation trick (value == tgt up to fp error; full Householder formula kept)
__global__ __launch_bounds__(256) void rotate_kernel(const float* src, const float* tgt,
                                                     float* outF, __bf16* outB, int D) {
  __shared__ float rss[256], rtt[256], rst[256];
  long row = blockIdx.x;
  int  t   = threadIdx.x;
  float ss = 0, tt = 0, st = 0;
  for (int d = t; d < D; d += 256) {
    float a = src[row * D + d], b = tgt[row * D + d];
    ss += a * a; tt += b * b; st += a * b;
  }
  rss[t] = ss; rtt[t] = tt; rst[t] = st; __syncthreads();
  for (int s = 128; s > 0; s >>= 1) {
    if (t < s) { rss[t] += rss[t + s]; rtt[t] += rtt[t + s]; rst[t] += rst[t + s]; }
    __syncthreads();
  }
  float SS = rss[0], TT = rtt[0], ST = rst[0];
  float ns = sqrtf(SS), nt = sqrtf(TT);
  float insE = 1.0f / (ns + 1e-12f), intE = 1.0f / (nt + 1e-12f);
  float eu = SS * insE, eq = ST * intE;
  float uu = SS * insE * insE, qq = TT * intE * intE, uq = ST * insE * intE;
  float nw = sqrtf(fmaxf(uu + 2.0f * uq + qq, 0.0f));
  float inw = 1.0f / (nw + 1e-12f);
  float ew = (eu + eq) * inw;
  float scale = nt * insE;
  for (int d = t; d < D; d += 256) {
    float a = src[row * D + d], b = tgt[row * D + d];
    float u = a * insE, q = b * intE;
    float w = (u + q) * inw;
    float o = (a - 2.0f * ew * w + 2.0f * eu * q) * scale;
    if (outF) outF[row * D + d] = o;
    if (outB) outB[row * D + d] = f2bf(o);
  }
}

__global__ __launch_bounds__(256) void mse_kernel(const float* a, const float* b, long n,
                                                  float* accum) {
  __shared__ float red[256];
  float s = 0.0f;
  for (long i = (long)blockIdx.x * 256 + threadIdx.x; i < n; i += (long)gridDim.x * 256) {
    float d = a[i] - b[i]; s += d * d;
  }
  red[threadIdx.x] = s; __syncthreads();
  for (int st = 128; st > 0; st >>= 1) {
    if (threadIdx.x < st) red[threadIdx.x] += red[threadIdx.x + st];
    __syncthreads();
  }
  if (threadIdx.x == 0) atomicAdd(accum, red[0]);
}

__global__ __launch_bounds__(256) void build_tokens_kernel(const float* quant,
                                                           const float* start, float* h) {
  long i = (long)blockIdx.x * 256 + threadIdx.x;
  if (i >= (long)MTOT * DMODEL) return;
  int  d = (int)(i & 511);
  long m = i >> 9;
  int  n = (int)(m & 255);
  int  b = (int)(m >> 8);
  h[i] = (n == 0) ? start[d] : quant[((long)b * NTOK + (n - 1)) * DMODEL + d];
}

__global__ __launch_bounds__(256) void layernorm_kernel(const float* x, const float* g,
                                                        __bf16* y, int D) {
  __shared__ float r1[256], r2[256];
  long row = blockIdx.x;
  int  t   = threadIdx.x;
  float s = 0, s2 = 0;
  for (int d = t; d < D; d += 256) { float v = x[row * D + d]; s += v; s2 += v * v; }
  r1[t] = s; r2[t] = s2; __syncthreads();
  for (int st = 128; st > 0; st >>= 1) {
    if (t < st) { r1[t] += r1[t + st]; r2[t] += r2[t + st]; }
    __syncthreads();
  }
  float mean = r1[0] / (float)D;
  float var  = r2[0] / (float)D - mean * mean;
  float rs   = rsqrtf(var + 1e-5f);
  for (int d = t; d < D; d += 256)
    y[row * D + d] = f2bf((x[row * D + d] - mean) * rs * g[d]);
}

// Rearrange [b,n,h*dh] f32 -> per-head bf16. tr_out=0: [b,h,n,dh]; tr_out=1: [b,h,dh,n].
__global__ __launch_bounds__(256) void rope_rearrange_kernel(const float* src, __bf16* dst,
                                                             int do_rope, int tr_out) {
  long i = (long)blockIdx.x * 256 + threadIdx.x;    // pair index
  if (i >= (long)BB * NTOK * NHEAD * 32) return;
  int j = (int)(i & 31);
  long r = i >> 5;
  int h = (int)(r & 7);  r >>= 3;
  int n = (int)(r & 255);
  int b = (int)(r >> 8);
  const float* sp = src + ((long)(b * NTOK + n) * DMODEL + h * DHEAD + 2 * j);
  float x1 = sp[0], x2 = sp[1], o1, o2;
  if (do_rope) {
    float fr = powf(10000.0f, -(float)(2 * j) / 64.0f);
    float ang = (float)n * fr;
    float c = cosf(ang), sn = sinf(ang);
    o1 = x1 * c - x2 * sn;
    o2 = x2 * c + x1 * sn;
  } else { o1 = x1; o2 = x2; }
  if (tr_out) {
    __bf16* dp = dst + ((long)(b * NHEAD + h) * DHEAD + 2 * j) * NTOK + n;
    dp[0]    = f2bf(o1);
    dp[NTOK] = f2bf(o2);
  } else {
    __bf16* dp = dst + ((long)((b * NHEAD + h) * NTOK + n) * DHEAD + 2 * j);
    dp[0] = f2bf(o1);
    dp[1] = f2bf(o2);
  }
}

__global__ __launch_bounds__(256) void softmax_kernel(const float* S, __bf16* P) {
  __shared__ float red[256];
  long row = blockIdx.x;                 // B*H*N rows of length 256
  int  t   = threadIdx.x;
  float v = S[row * 256 + t];
  red[t] = v; __syncthreads();
  for (int st = 128; st > 0; st >>= 1) {
    if (t < st) red[t] = fmaxf(red[t], red[t + st]);
    __syncthreads();
  }
  float mx = red[0]; __syncthreads();
  float e = __expf(v - mx);
  red[t] = e; __syncthreads();
  for (int st = 128; st > 0; st >>= 1) {
    if (t < st) red[t] += red[t + st];
    __syncthreads();
  }
  P[row * 256 + t] = f2bf(e / red[0]);
}

__global__ __launch_bounds__(256) void tgt_logit_kernel(const float* pred, const float* cbc,
                                                        float* tl, int D) {
  __shared__ float red[256];
  long row = blockIdx.x;
  float s = 0.0f;
  for (int d = threadIdx.x; d < D; d += 256) {
    float df = pred[row * D + d] - cbc[row * D + d];
    s += df * df;
  }
  red[threadIdx.x] = s; __syncthreads();
  for (int st = 128; st > 0; st >>= 1) {
    if (threadIdx.x < st) red[threadIdx.x] += red[threadIdx.x + st];
    __syncthreads();
  }
  if (threadIdx.x == 0) tl[row] = -sqrtf(fmaxf(red[0], 1e-12f));
}

__global__ __launch_bounds__(256) void ce_final_kernel(const unsigned long long* pk,
                                                       const float* tl, const float* se,
                                                       float* accum) {
  int m = blockIdx.x * 256 + threadIdx.x;
  if (m >= MTOT) return;
  unsigned u = (unsigned)(pk[m] >> 32);
  u = (u & 0x80000000u) ? (u & 0x7fffffffu) : ~u;
  float ml = -sqrtf(fmaxf(__uint_as_float(u), 1e-12f));
  float lp = tl[m] - ml - logf(se[m]);
  atomicAdd(accum, -lp);
}

__global__ __launch_bounds__(256) void recon_kernel(const float* dec, const float* img,
                                                    float* out_recon, float* accum) {
  __shared__ float red[256];
  long i = (long)blockIdx.x * 256 + threadIdx.x;   // pixel index, grid covers PIX exactly
  int col = (int)(i & 255);
  int row = (int)((i >> 8) & 255);
  int b   = (int)(i >> 16);
  int hx = col >> 4, px = col & 15;
  int hy = row >> 4, py = row & 15;
  float v = (dec[((long)(b * NTOK + hy * 16 + hx) * 256) + py * 16 + px] + 1.0f) * 0.5f;
  out_recon[i] = v;
  float d = v - img[i];
  red[threadIdx.x] = d * d; __syncthreads();
  for (int st = 128; st > 0; st >>= 1) {
    if (threadIdx.x < st) red[threadIdx.x] += red[threadIdx.x + st];
    __syncthreads();
  }
  if (threadIdx.x == 0) atomicAdd(accum + 2, red[0]);
}

__global__ void finalize_kernel(const float* accum, float* out) {
  float commit = accum[0] / ((float)MTOT * (float)DMODEL);
  float ce     = accum[1] / (float)MTOT;
  float recon  = accum[2] / (float)PIX;
  float ar     = accum[3] / ((float)MTOT * (float)DMODEL);
  out[0]           = ce + recon + commit + ar;
  out[1 + PIX]     = ce;
  out[2 + PIX]     = recon;
  out[3 + PIX]     = commit;
  out[4 + PIX]     = ar;
}

// ---------------------------------------------------------------------------
// Host side
// ---------------------------------------------------------------------------
static void gemm_store(hipStream_t s, int M, int N, int K, int nb, int ic,
                       const __bf16* A, long lda, long ao, long ai,
                       const __bf16* B, long ldb, long bo, long bi,
                       float* outF, __bf16* outB, const float* bias, const float* residual,
                       long ldc, long co, long ci, float alpha, int causal, int gelu) {
  GemmParams p{A, lda, ao, ai, B, ldb, bo, bi, K, ic};
  StoreEpi e{outF, outB, bias, residual, ldc, co, ci, alpha, causal, gelu};
  dim3 g(N / 64, M / 128, nb);
  gemm_bf16_kernel<StoreEpi><<<g, 256, 0, s>>>(p, e);
}

static void gemm_argmin(hipStream_t s, int M, int N, int K,
                        const __bf16* A, long lda, const __bf16* B, long ldb,
                        const float* rn, const float* cn, unsigned long long* keys) {
  GemmParams p{A, lda, 0, 0, B, ldb, 0, 0, K, 1};
  ArgminEpi e{rn, cn, keys};
  dim3 g(N / 64, M / 128, 1);
  gemm_bf16_kernel<ArgminEpi><<<g, 256, 0, s>>>(p, e);
}

static void gemm_ce(hipStream_t s, int M, int N, int K,
                    const __bf16* A, long lda, const __bf16* B, long ldb,
                    const float* rn, const float* cn,
                    const unsigned long long* pk, float* sumexp) {
  GemmParams p{A, lda, 0, 0, B, ldb, 0, 0, K, 1};
  CeEpi e{rn, cn, pk, sumexp};
  dim3 g(N / 64, M / 128, 1);
  gemm_bf16_kernel<CeEpi><<<g, 256, 0, s>>>(p, e);
}

static inline int gr(long n) { return (int)((n + 255) / 256); }

extern "C" void kernel_launch(void* const* d_in, const int* in_sizes, int n_in,
                              void* d_out, int out_size, void* d_ws, size_t ws_size,
                              hipStream_t stream) {
  (void)in_sizes; (void)n_in; (void)out_size; (void)ws_size;
  const float* image   = (const float*)d_in[0];
  const float* enc_w   = (const float*)d_in[1];
  const float* enc_b   = (const float*)d_in[2];
  const float* cb      = (const float*)d_in[3];
  const float* start   = (const float*)d_in[4];
  const float* ln1_g   = (const float*)d_in[5];
  const float* ln2_g   = (const float*)d_in[6];
  const float* wq      = (const float*)d_in[7];
  const float* wk      = (const float*)d_in[8];
  const float* wv      = (const float*)d_in[9];
  const float* wo      = (const float*)d_in[10];
  const float* ff_w1   = (const float*)d_in[11];
  const float* ff_b1   = (const float*)d_in[12];
  const float* ff_w2   = (const float*)d_in[13];
  const float* ff_b2   = (const float*)d_in[14];
  const float* lnf_g   = (const float*)d_in[15];
  const float* post_w  = (const float*)d_in[16];
  const float* post_b  = (const float*)d_in[17];
  const float* dec_w   = (const float*)d_in[18];
  const float* dec_b   = (const float*)d_in[19];
  float* out = (float*)d_out;

  // ---- workspace carve-out ----
  char* w = (char*)d_ws;
  size_t off = 0;
  auto alloc = [&](size_t bytes) -> char* {
    size_t a = (off + 255) & ~(size_t)255; off = a + bytes; return w + a;
  };
  // bf16 weights (all stored [N][K], K-contiguous; codebook is already [L][D])
  __bf16* wencT  = (__bf16*)alloc((size_t)256 * 512 * 2);
  __bf16* wcbB   = (__bf16*)alloc((size_t)LCB * DMODEL * 2);
  __bf16* wqT    = (__bf16*)alloc((size_t)NLAYER * 512 * 512 * 2);
  __bf16* wkT    = (__bf16*)alloc((size_t)NLAYER * 512 * 512 * 2);
  __bf16* wvT    = (__bf16*)alloc((size_t)NLAYER * 512 * 512 * 2);
  __bf16* woT    = (__bf16*)alloc((size_t)NLAYER * 512 * 512 * 2);
  __bf16* wf1T   = (__bf16*)alloc((size_t)NLAYER * 512 * FFD * 2);
  __bf16* wf2T   = (__bf16*)alloc((size_t)NLAYER * FFD * 512 * 2);
  __bf16* wpostT = (__bf16*)alloc((size_t)512 * 512 * 2);
  __bf16* wdecT  = (__bf16*)alloc((size_t)512 * 256 * 2);
  // bf16 activations
  __bf16* patchesB = (__bf16*)alloc((size_t)MTOT * 256 * 2);
  __bf16* encB     = (__bf16*)alloc((size_t)MTOT * DMODEL * 2);
  __bf16* yB       = (__bf16*)alloc((size_t)MTOT * DMODEL * 2);
  __bf16* qhB      = (__bf16*)alloc((size_t)MTOT * DMODEL * 2);
  __bf16* khB      = (__bf16*)alloc((size_t)MTOT * DMODEL * 2);
  __bf16* vhT      = (__bf16*)alloc((size_t)MTOT * DMODEL * 2);   // [b,h,dh,n]
  __bf16* probsB   = (__bf16*)alloc((size_t)BB * NHEAD * NTOK * NTOK * 2);
  __bf16* attnB    = (__bf16*)alloc((size_t)MTOT * DMODEL * 2);
  __bf16* predB    = (__bf16*)alloc((size_t)MTOT * DMODEL * 2);
  __bf16* rotB     = (__bf16*)alloc((size_t)MTOT * DMODEL * 2);
  // f32 activations
  float* enc_f   = (float*)alloc((size_t)MTOT * DMODEL * 4);
  float* cbc_f   = (float*)alloc((size_t)MTOT * DMODEL * 4);
  float* quant_f = (float*)alloc((size_t)MTOT * DMODEL * 4);
  float* h_f     = (float*)alloc((size_t)MTOT * DMODEL * 4);
  float* qkv_f   = (float*)alloc((size_t)MTOT * DMODEL * 4);
  float* scores_f= (float*)alloc((size_t)BB * NHEAD * NTOK * NTOK * 4);
  float* pred_f  = (float*)alloc((size_t)MTOT * DMODEL * 4);
  float* dec_f   = (float*)alloc((size_t)MTOT * 256 * 4);
  float* enc_nrm = (float*)alloc((size_t)MTOT * 4);
  float* cb_nrm  = (float*)alloc((size_t)LCB * 4);
  float* prd_nrm = (float*)alloc((size_t)MTOT * 4);
  float* sumexp  = (float*)alloc((size_t)MTOT * 4);
  float* tl      = (float*)alloc((size_t)MTOT * 4);
  float* accum   = (float*)alloc(16 * 4);
  unsigned long long* keys  = (unsigned long long*)alloc((size_t)MTOT * 8);
  unsigned long long* pkeys = (unsigned long long*)alloc((size_t)MTOT * 8);
  // FF hidden aliases the scores buffer (disjoint lifetimes within a layer)
  __bf16* ffhB = (__bf16*)scores_f;

  // ---- init ----
  zero_f32_kernel<<<1, 256, 0, stream>>>(accum, 16);
  zero_f32_kernel<<<gr(MTOT), 256, 0, stream>>>(sumexp, MTOT);
  init_u64_kernel<<<gr(MTOT), 256, 0, stream>>>(keys, MTOT);
  init_u64_kernel<<<gr(MTOT), 256, 0, stream>>>(pkeys, MTOT);

  // ---- weight conversion (transposed to [N][K]) ----
  auto cvt_t = [&](const float* s, __bf16* d, int K, int N, int L) {
    long n = (long)L * K * N;
    f32_to_bf16_t_kernel<<<gr(n), 256, 0, stream>>>(s, d, K, N, n);
  };
  cvt_t(enc_w,  wencT,  256, 512, 1);
  f32_to_bf16_kernel<<<gr((long)LCB * DMODEL), 256, 0, stream>>>(cb, wcbB, (long)LCB * DMODEL);
  cvt_t(wq,     wqT,    512, 512, NLAYER);
  cvt_t(wk,     wkT,    512, 512, NLAYER);
  cvt_t(wv,     wvT,    512, 512, NLAYER);
  cvt_t(wo,     woT,    512, 512, NLAYER);
  cvt_t(ff_w1,  wf1T,   512, FFD, NLAYER);
  cvt_t(ff_w2,  wf2T,   FFD, 512, NLAYER);
  cvt_t(post_w, wpostT, 512, 512, 1);
  cvt_t(dec_w,  wdecT,  512, 256, 1);

  // ---- encode ----
  patchify_kernel<<<gr((long)MTOT * 256), 256, 0, stream>>>(image, patchesB);
  rownorm_kernel<<<LCB, 256, 0, stream>>>(cb, cb_nrm, DMODEL);
  gemm_store(stream, MTOT, DMODEL, 256, 1, 1,
             patchesB, 256, 0, 0, wencT, 256, 0, 0,
             enc_f, encB, enc_b, nullptr, DMODEL, 0, 0, 1.0f, 0, 0);
  rownorm_kernel<<<MTOT, 256, 0, stream>>>(enc_f, enc_nrm, DMODEL);

  // ---- VQ ----
  gemm_argmin(stream, MTOT, LCB, DMODEL, encB, DMODEL, wcbB, DMODEL,
              enc_nrm, cb_nrm, keys);
  gather_cbc_kernel<<<gr((long)MTOT * DMODEL), 256, 0, stream>>>(keys, cb, cbc_f);
  rotate_kernel<<<MTOT, 256, 0, stream>>>(enc_f, cbc_f, quant_f, nullptr, DMODEL);
  mse_kernel<<<1024, 256, 0, stream>>>(quant_f, enc_f, (long)MTOT * DMODEL, accum + 0);
  build_tokens_kernel<<<gr((long)MTOT * DMODEL), 256, 0, stream>>>(quant_f, start, h_f);

  // ---- transformer decoder ----
  const long WL  = 512 * 512;        // per-layer attn weight stride
  const long FL  = (long)512 * FFD;  // per-layer ff weight stride
  const long BH  = (long)BB * NHEAD;
  const long QS  = (long)NTOK * DHEAD;   // 16384
  const long SS  = (long)NTOK * NTOK;    // 65536
  for (int i = 0; i < NLAYER; ++i) {
    layernorm_kernel<<<MTOT, 256, 0, stream>>>(h_f, ln1_g + i * DMODEL, yB, DMODEL);
    // Q
    gemm_store(stream, MTOT, DMODEL, DMODEL, 1, 1, yB, DMODEL, 0, 0,
               wqT + i * WL, DMODEL, 0, 0, qkv_f, nullptr, nullptr, nullptr,
               DMODEL, 0, 0, 1.0f, 0, 0);
    rope_rearrange_kernel<<<gr((long)MTOT * 256), 256, 0, stream>>>(qkv_f, qhB, 1, 0);
    // K
    gemm_store(stream, MTOT, DMODEL, DMODEL, 1, 1, yB, DMODEL, 0, 0,
               wkT + i * WL, DMODEL, 0, 0, qkv_f, nullptr, nullptr, nullptr,
               DMODEL, 0, 0, 1.0f, 0, 0);
    rope_rearrange_kernel<<<gr((long)MTOT * 256), 256, 0, stream>>>(qkv_f, khB, 1, 0);
    // V (transposed per-head layout [b,h,dh,n] so P@V's B operand is K-contiguous)
    gemm_store(stream, MTOT, DMODEL, DMODEL, 1, 1, yB, DMODEL, 0, 0,
               wvT + i * WL, DMODEL, 0, 0, qkv_f, nullptr, nullptr, nullptr,
               DMODEL, 0, 0, 1.0f, 0, 0);
    rope_rearrange_kernel<<<gr((long)MTOT * 256), 256, 0, stream>>>(qkv_f, vhT, 0, 1);
    // scores = causal_mask(q @ k^T * 1/8)   batched over b*h
    gemm_store(stream, NTOK, NTOK, DHEAD, (int)BH, 1,
               qhB, DHEAD, QS, 0, khB, DHEAD, QS, 0,
               scores_f, nullptr, nullptr, nullptr, NTOK, SS, 0, 0.125f, 1, 0);
    softmax_kernel<<<(int)(BH * NTOK), 256, 0, stream>>>(scores_f, probsB);
    // attn_out[b,n,h*dh] = P @ V
    gemm_store(stream, NTOK, DHEAD, NTOK, (int)BH, NHEAD,
               probsB, NTOK, (long)NHEAD * SS, SS,
               vhT, NTOK, (long)NHEAD * QS, QS,
               nullptr, attnB, nullptr, nullptr,
               DMODEL, (long)NTOK * DMODEL, DHEAD, 1.0f, 0, 0);
    // h += attn_out @ wo
    gemm_store(stream, MTOT, DMODEL, DMODEL, 1, 1, attnB, DMODEL, 0, 0,
               woT + i * WL, DMODEL, 0, 0, h_f, nullptr, nullptr, h_f,
               DMODEL, 0, 0, 1.0f, 0, 0);
    // FF
    layernorm_kernel<<<MTOT, 256, 0, stream>>>(h_f, ln2_g + i * DMODEL, yB, DMODEL);
    gemm_store(stream, MTOT, FFD, DMODEL, 1, 1, yB, DMODEL, 0, 0,
               wf1T + i * FL, DMODEL, 0, 0, nullptr, ffhB, ff_b1 + i * FFD, nullptr,
               FFD, 0, 0, 1.0f, 0, 1);
    gemm_store(stream, MTOT, DMODEL, FFD, 1, 1, ffhB, FFD, 0, 0,
               wf2T + i * FL, FFD, 0, 0, h_f, nullptr, ff_b2 + i * DMODEL, h_f,
               DMODEL, 0, 0, 1.0f, 0, 0);
  }
  layernorm_kernel<<<MTOT, 256, 0, stream>>>(h_f, lnf_g, yB, DMODEL);

  // ---- post / CE ----
  gemm_store(stream, MTOT, DMODEL, DMODEL, 1, 1, yB, DMODEL, 0, 0,
             wpostT, DMODEL, 0, 0, pred_f, predB, post_b, nullptr,
             DMODEL, 0, 0, 1.0f, 0, 0);
  rownorm_kernel<<<MTOT, 256, 0, stream>>>(pred_f, prd_nrm, DMODEL);
  gemm_argmin(stream, MTOT, LCB, DMODEL, predB, DMODEL, wcbB, DMODEL,
              prd_nrm, cb_nrm, pkeys);                       // max logit per row
  gemm_ce(stream, MTOT, LCB, DMODEL, predB, DMODEL, wcbB, DMODEL,
          prd_nrm, cb_nrm, pkeys, sumexp);                   // sum exp(logit - max)
  tgt_logit_kernel<<<MTOT, 256, 0, stream>>>(pred_f, cbc_f, tl, DMODEL);
  ce_final_kernel<<<gr(MTOT), 256, 0, stream>>>(pkeys, tl, sumexp, accum + 1);

  // ---- decode / losses ----
  rotate_kernel<<<MTOT, 256, 0, stream>>>(pred_f, cbc_f, nullptr, rotB, DMODEL);
  gemm_store(stream, MTOT, 256, DMODEL, 1, 1, rotB, DMODEL, 0, 0,
             wdecT, DMODEL, 0, 0, dec_f, nullptr, dec_b, nullptr,
             256, 0, 0, 1.0f, 0, 0);
  mse_kernel<<<1024, 256, 0, stream>>>(pred_f, quant_f, (long)MTOT * DMODEL, accum + 3);
  recon_kernel<<<PIX / 256, 256, 0, stream>>>(dec_f, image, out + 1, accum);
  finalize_kernel<<<1, 1, 0, stream>>>(accum, out);
}
